// ResidualBlock_48146583388640
// MI455X (gfx1250) — matrix operand order repaired and checked by reading, not compiled, not run on hardware
//
#include <hip/hip_runtime.h>

#define HIDDEN 64
#define T_LEN  32768
#define NSKIP  256
#define TT     64     // time steps per block
#define HALO   64     // dilation lookback

typedef __attribute__((ext_vector_type(16))) __bf16 v16bf;
typedef __attribute__((ext_vector_type(8)))  float  v8f;

union Frag32B { uint4 q[2]; v16bf v; };

__device__ __forceinline__ unsigned short f2bf(float x) {
  unsigned int u = __float_as_uint(x);
  u += 0x7FFFu + ((u >> 16) & 1u);          // round-to-nearest-even
  return (unsigned short)(u >> 16);
}

__device__ __forceinline__ v8f wmma_bf16(v16bf a, v16bf b, v8f c) {
  // (neg_a, A, neg_b, B, c_mod, C, reuse_a, reuse_b)
  return __builtin_amdgcn_wmma_f32_16x16x32_bf16(false, a, false, b, (short)0, c,
                                                 false, false);
}

// A-matrix (16xK bf16): lane L = row M=L%16; 16 contiguous K at 2*kb, kb=(L<16?0:8)
__device__ __forceinline__ v16bf load_fragA(const unsigned short* row_ptr, int kb) {
  Frag32B f;
  f.q[0] = *(const uint4*)(row_ptr + 2 * kb);
  f.q[1] = *(const uint4*)(row_ptr + 2 * kb + 8);
  return f.v;
}

// B-matrix (32x16 bf16): lane L = col N=L%16; 16 contiguous K at kb16=(L<16?0:16)
__device__ __forceinline__ v16bf load_fragB(const unsigned short* p) {
  Frag32B f;
  f.q[0] = *(const uint4*)(p);
  f.q[1] = *(const uint4*)(p + 8);
  return f.v;
}

__global__ __launch_bounds__(256) void
wavenet_block_kernel(const float* __restrict__ sig,
                     const float* __restrict__ cond,
                     const float* __restrict__ Wdil,
                     const float* __restrict__ bdil,
                     const float* __restrict__ Wout,
                     const float* __restrict__ bout,
                     const float* __restrict__ Wskip,
                     const float* __restrict__ bskip,
                     float* __restrict__ out,
                     float* __restrict__ skp)
{
  // 58 KB static LDS, phase-reused:
  //   [0,32768)      : A_dil[128][128]  -> later A_skip[256][64]
  //   [32768,49152)  : sig tile [128][64] (transposed) -> later A_out[64][64]
  //   [49152,57344)  : gated [64][64]
  //   [57344,59136)  : biases (f32)
  __shared__ __align__(16) unsigned char smem[59136];
  unsigned short* sW   = (unsigned short*)(smem);
  unsigned short* sSig = (unsigned short*)(smem + 32768);
  unsigned short* sAo  = sSig;
  unsigned short* sG   = (unsigned short*)(smem + 49152);
  float* sBd = (float*)(smem + 57344);   // 128
  float* sBs = sBd + 128;                // 256
  float* sBo = sBs + 256;                // 64

  const int tid  = threadIdx.x;
  const int wave = tid >> 5;
  const int lane = tid & 31;
  const int hi   = lane >> 4;
  const int lm   = lane & 15;

  const int b    = blockIdx.x >> 9;      // T_LEN/TT = 512 tiles per batch
  const int tile = blockIdx.x & 511;
  const int t0   = tile * TT;

  // ---- stage: signal tile (transposed, with causal halo), conv weights, biases ----
  for (int idx = tid; idx < (TT + HALO) * HIDDEN; idx += 256) {
    int tl = idx & 127;                   // lanes walk t -> coalesced global reads
    int c  = idx >> 7;
    int tg = t0 - HALO + tl;
    float v = (tg >= 0) ? sig[((size_t)b * HIDDEN + c) * T_LEN + tg] : 0.0f;
    sSig[tl * HIDDEN + c] = f2bf(v);
  }
  for (int idx = tid; idx < 128 * 128; idx += 256) {
    int o = idx >> 7, k = idx & 127;      // A_conv[o][k] ; k<64 = tap0 (t-64), k>=64 = tap1 (t)
    sW[idx] = f2bf(Wdil[((size_t)o * HIDDEN + (k & 63)) * 2 + (k >> 6)]);
  }
  if (tid < 128) sBd[tid] = bdil[tid];
  sBs[tid] = bskip[tid];
  if (tid < 64) sBo[tid] = bout[tid];
  __syncthreads();

  // ---- phase 1: dilated conv as 128xN GEMM with K=128, then gate -> sG ----
  for (int i = 0; i < 2; ++i) {
    int task = wave * 2 + i;              // 16 tasks / 8 waves
    int m = task & 3;                     // row-tile pair: m (tanh) / m+4 (sigmoid)
    int n = task >> 2;                    // 16-wide time column tile
    v8f accT = {}, accS = {};
    for (int kk = 0; kk < 4; ++kk) {
      int tl = ((kk < 2) ? 0 : HALO) + n * 16 + lm;   // tap0 reads t-64 (halo rows)
      int c0 = (kk & 1) * 32 + hi * 16;
      v16bf bf = load_fragB(sSig + tl * HIDDEN + c0);
      v16bf aT = load_fragA(sW + (16 * m + lm) * 128 + kk * 32, hi * 8);
      v16bf aS = load_fragA(sW + (64 + 16 * m + lm) * 128 + kk * 32, hi * 8);
      accT = wmma_bf16(aT, bf, accT);
      accS = wmma_bf16(aS, bf, accS);
    }
    int tg = t0 + n * 16 + lm;
    const float* condB = cond + (size_t)b * 128 * T_LEN;
    unsigned short h[8];
    for (int r = 0; r < 8; ++r) {
      int oT = 16 * m + r + 8 * hi;
      int oS = oT + HIDDEN;
      float xt = accT[r] + sBd[oT] + condB[(size_t)oT * T_LEN + tg];
      float xs = accS[r] + sBd[oS] + condB[(size_t)oS * T_LEN + tg];
      float e2 = __expf(-2.0f * xt);
      float th = (1.0f - e2) / (1.0f + e2);             // tanh
      float sg = 1.0f / (1.0f + __expf(-xs));           // sigmoid
      h[r] = f2bf(th * sg);
    }
    uint4 q;
    q.x = (unsigned)h[0] | ((unsigned)h[1] << 16);
    q.y = (unsigned)h[2] | ((unsigned)h[3] << 16);
    q.z = (unsigned)h[4] | ((unsigned)h[5] << 16);
    q.w = (unsigned)h[6] | ((unsigned)h[7] << 16);
    *(uint4*)(sG + (n * 16 + lm) * HIDDEN + 16 * m + 8 * hi) = q;   // sG[t][c]
  }
  __syncthreads();

  // ---- stage skip/out weights into the (now dead) regions ----
  for (int idx = tid; idx < NSKIP * HIDDEN; idx += 256)  sW[idx]  = f2bf(Wskip[idx]);
  for (int idx = tid; idx < HIDDEN * HIDDEN; idx += 256) sAo[idx] = f2bf(Wout[idx]);
  __syncthreads();

  // ---- phase 3: skip (256x64) and out (64x64) GEMMs over gated ----
  for (int task = wave; task < 80; task += 8) {          // wave-uniform
    const bool isSkip = (task < 64);
    int mrow, n;
    if (isSkip) { mrow = task & 15; n = task >> 4; }
    else        { int tt = task - 64; mrow = tt & 3; n = tt >> 2; }
    const unsigned short* A = isSkip ? sW : sAo;
    v8f acc = {};
    for (int kk = 0; kk < 2; ++kk) {
      v16bf bf = load_fragB(sG + (n * 16 + lm) * HIDDEN + kk * 32 + hi * 16);
      v16bf a  = load_fragA(A + (16 * mrow + lm) * HIDDEN + kk * 32, hi * 8);
      acc = wmma_bf16(a, bf, acc);
    }
    int tg = t0 + n * 16 + lm;
    if (isSkip) {
      for (int r = 0; r < 8; ++r) {
        int o = 16 * mrow + r + 8 * hi;
        skp[((size_t)b * NSKIP + o) * T_LEN + tg] = acc[r] + sBs[o];
      }
    } else {
      for (int r = 0; r < 8; ++r) {
        int o = 16 * mrow + r + 8 * hi;
        size_t off = ((size_t)b * HIDDEN + o) * T_LEN + tg;
        out[off] = acc[r] + sBo[o] + sig[off];           // + residual
      }
    }
  }
}

extern "C" void kernel_launch(void* const* d_in, const int* in_sizes, int n_in,
                              void* d_out, int out_size, void* d_ws, size_t ws_size,
                              hipStream_t stream) {
  const float* sig   = (const float*)d_in[0];
  const float* cond  = (const float*)d_in[1];
  const float* Wdil  = (const float*)d_in[2];
  const float* bdil  = (const float*)d_in[3];
  const float* Wout  = (const float*)d_in[4];
  const float* bout  = (const float*)d_in[5];
  const float* Wskip = (const float*)d_in[6];
  const float* bskip = (const float*)d_in[7];
  float* out = (float*)d_out;                               // [8,64,32768] first
  float* skp = out + (size_t)8 * HIDDEN * T_LEN;            // then [8,256,32768]
  (void)in_sizes; (void)n_in; (void)out_size; (void)d_ws; (void)ws_size;
  dim3 grid(8 * (T_LEN / TT));                              // 4096 blocks
  wavenet_block_kernel<<<grid, 256, 0, stream>>>(sig, cond, Wdil, bdil,
                                                 Wout, bout, Wskip, bskip,
                                                 out, skp);
}
